// CommsLSTMAgent_11416023072921
// MI455X (gfx1250) — matrix operand order repaired and checked by reading, not compiled, not run on hardware
//
#include <hip/hip_runtime.h>
#include <cstddef>

// ---------------------------------------------------------------------------
// CommsLSTMAgent forward for MI455X (gfx1250), wave32, WMMA f16 -> f32 acc.
// GEMM: 32x32 tile per wave (4 wmma / K-step), software-pipelined K loop
// (two fragment register sets, loads for k+32 in flight under wmma on k).
// One 8-wave block covers 256 rows x 32 cols so all waves share B fragments.
// ---------------------------------------------------------------------------

typedef _Float16 f16;
typedef _Float16 half8 __attribute__((ext_vector_type(8)));
typedef _Float16 v16h  __attribute__((ext_vector_type(16)));
typedef float    v8f   __attribute__((ext_vector_type(8)));

union H16 { v16h v; half8 h[2]; };

#define T_STEPS 16
#define B_BATCH 128
#define N_TOT   (T_STEPS * B_BATCH)   // 2048
#define HID     512
#define LIN_IN_PAD 576                // 550 padded to /32
#define WPB     8                     // waves per block

__device__ __forceinline__ float sigm(float x) { return 1.0f / (1.0f + __expf(-x)); }

#define WMMA_F16(a, b, c) \
    __builtin_amdgcn_wmma_f32_16x16x32_f16(false, (a).v, false, (b).v, (short)0, (c), false, false)

// ---------------------------------------------------------------------------
// C[M,Nc] = A[M,K](f16 rm) * B[Nc,K](f16 rm)^T
// Requirements: M % 32 == 0, Nc % 32 == 0, K % 64 == 0.
// Grid: x = ceil(M/256) (8 waves * 32 rows), y = Nc/32.
// Epilogue: optional bias[col], act 0=none 1=relu 2=tanh, f16 or f32 out.
// SP>1: conv NCHW scatter  idx = ((m/SP)*NCtot + col)*SP + m%SP
// else: idx = m*ldc + col + col_off.
// ---------------------------------------------------------------------------
__global__ void __launch_bounds__(256)
gemm_wmma_f16(const f16* __restrict__ A, int lda,
              const f16* __restrict__ B, int ldb,
              int M, int Nc, int K,
              const float* __restrict__ bias, int act,
              float* __restrict__ outF, f16* __restrict__ outH,
              int ldc, int col_off, int SP, int NCtot)
{
    const int lane = threadIdx.x & 31;
    const int wav  = threadIdx.x >> 5;
    const int m0   = (blockIdx.x * WPB + wav) * 32;
    if (m0 >= M) return;
    const int n0 = blockIdx.y * 32;
    const int r  = lane & 15;
    // A fragment K chunks (ISA 16-bit A layout): lanes 0-15 -> {k..k+7, k+16..23},
    // lanes 16-31 -> {k+8..15, k+24..31}. B fragment: same K chunks as A.
    const int ak = (lane & 16) ? 8 : 0;
    const int bk = (lane & 16) ? 8 : 0;

    const f16* Ar0 = A + (size_t)(m0 + r) * lda + ak;
    const f16* Ar1 = A + (size_t)(m0 + 16 + r) * lda + ak;
    const f16* Br0 = B + (size_t)(n0 + r) * ldb + bk;
    const f16* Br1 = B + (size_t)(n0 + 16 + r) * ldb + bk;

    v8f acc00 = {0.f,0.f,0.f,0.f,0.f,0.f,0.f,0.f};
    v8f acc01 = acc00, acc10 = acc00, acc11 = acc00;

    H16 a0A, a1A, b0A, b1A;     // fragment set A
    H16 a0B, a1B, b0B, b1B;     // fragment set B (pipeline partner)

    auto load_set = [&](H16& a0, H16& a1, H16& b0, H16& b1, int k0) {
        a0.h[0] = *(const half8*)(Ar0 + k0);
        a0.h[1] = *(const half8*)(Ar0 + k0 + 16);
        a1.h[0] = *(const half8*)(Ar1 + k0);
        a1.h[1] = *(const half8*)(Ar1 + k0 + 16);
        b0.h[0] = *(const half8*)(Br0 + k0);
        b0.h[1] = *(const half8*)(Br0 + k0 + 16);
        b1.h[0] = *(const half8*)(Br1 + k0);
        b1.h[1] = *(const half8*)(Br1 + k0 + 16);
    };
    auto mm4 = [&](H16& a0, H16& a1, H16& b0, H16& b1) {
        acc00 = WMMA_F16(a0, b0, acc00);
        acc01 = WMMA_F16(a0, b1, acc01);
        acc10 = WMMA_F16(a1, b0, acc10);
        acc11 = WMMA_F16(a1, b1, acc11);
    };

    load_set(a0A, a1A, b0A, b1A, 0);
    for (int k0 = 0; k0 < K; k0 += 64) {
        load_set(a0B, a1B, b0B, b1B, k0 + 32);           // in flight under mm4(A)
        if (k0 + 128 < K) {
            __builtin_prefetch(Ar0 + k0 + 128, 0, 1);    // global_prefetch_b8
            __builtin_prefetch(Ar1 + k0 + 128, 0, 1);
        }
        mm4(a0A, a1A, b0A, b1A);
        if (k0 + 64 < K)
            load_set(a0A, a1A, b0A, b1A, k0 + 64);       // in flight under mm4(B)
        mm4(a0B, a1B, b0B, b1B);
    }

    // C layout: VGPR v -> row v (lanes 0-15) / v+8 (lanes 16-31); col = lane%16.
    const int mh  = (lane & 16) ? 8 : 0;
    const int c0  = n0 + r, c1 = n0 + 16 + r;
    const float bv0 = bias ? bias[c0] : 0.0f;
    const float bv1 = bias ? bias[c1] : 0.0f;

    auto emit = [&](int m, int col, float x, float bv) {
        x += bv;
        if      (act == 1) x = fmaxf(x, 0.0f);
        else if (act == 2) x = tanhf(x);
        size_t idx;
        if (SP > 1) idx = ((size_t)(m / SP) * NCtot + (size_t)(col + col_off)) * SP + (m % SP);
        else        idx = (size_t)m * ldc + col + col_off;
        if (outH) outH[idx] = (f16)x;
        else      outF[idx] = x;
    };
#pragma unroll
    for (int v = 0; v < 8; ++v) {
        int mA = m0 + mh + v;        // rows of acc0*
        int mB = mA + 16;            // rows of acc1*
        emit(mA, c0, acc00[v], bv0);
        emit(mA, c1, acc01[v], bv1);
        emit(mB, c0, acc10[v], bv0);
        emit(mB, c1, acc11[v], bv1);
    }
}

// --------------------------- im2col (f16 staging) --------------------------
// conv1: x[N,84,84,3] (NHWC); torch transpose(1,3) -> h[n,c,a,b]=x[n,b,a,c]
// k = c*64 + i*8 + j  matches w1 flatten [o,c,i,j].
__global__ void im2col1(const float* __restrict__ x, f16* __restrict__ Ab,
                        int n_base, int nelem)
{
    int e = blockIdx.x * blockDim.x + threadIdx.x;
    if (e >= nelem) return;
    int k = e % 192, ml = e / 192;
    int n = n_base + ml / 400, s = ml % 400;
    int p = s / 20, q = s % 20;                 // p: W-pos, q: H-pos
    int c = k >> 6, rr = k & 63, i = rr >> 3, j = rr & 7;
    float v = x[((size_t)(n * 84 + (q * 4 + j)) * 84 + (p * 4 + i)) * 3 + c];
    Ab[e] = (f16)v;
}

// conv2: Y1[n,c,20,20] f16, 4x4 stride 2 -> k = c*16 + i*4 + j
__global__ void im2col2(const f16* __restrict__ Y1, f16* __restrict__ Ab,
                        int n_base, int nelem)
{
    int e = blockIdx.x * blockDim.x + threadIdx.x;
    if (e >= nelem) return;
    int k = e % 512, ml = e / 512;
    int n = n_base + ml / 81, s = ml % 81;
    int p = s / 9, q = s % 9;
    int c = k >> 4, rr = k & 15, i = rr >> 2, j = rr & 3;
    Ab[e] = Y1[((size_t)(n * 32 + c) * 20 + (p * 2 + i)) * 20 + (q * 2 + j)];
}

// conv3: Y2[n,c,9,9] f16, 3x3 stride 1 -> k = c*9 + i*3 + j
__global__ void im2col3(const f16* __restrict__ Y2, f16* __restrict__ Ab,
                        int n_base, int nelem)
{
    int e = blockIdx.x * blockDim.x + threadIdx.x;
    if (e >= nelem) return;
    int k = e % 576, ml = e / 576;
    int n = n_base + ml / 49, s = ml % 49;
    int p = s / 7, q = s % 7;
    int c = k / 9, rr = k % 9, i = rr / 3, j = rr % 3;
    Ab[e] = Y2[((size_t)(n * 64 + c) * 9 + (p + i)) * 9 + (q + j)];
}

// ------------------ weight cast fp32 -> f16 with zero padding ---------------
__global__ void cast_pad(const float* __restrict__ src, f16* __restrict__ dst,
                         int O, int K, int Opad, int Kp)
{
    int e = blockIdx.x * blockDim.x + threadIdx.x;
    if (e >= Opad * Kp) return;
    int o = e / Kp, k = e - o * Kp;
    dst[e] = (o < O && k < K) ? (f16)src[(size_t)o * K + k] : (f16)0.0f;
}

// ---------- LSTM-input auxiliary columns of XL[2048, 576] (cols 512..575) ---
__global__ void fill_aux(const float* __restrict__ la, const float* __restrict__ lr,
                         const int* __restrict__ msg, const float* __restrict__ ct,
                         const float* __restrict__ tte, f16* __restrict__ XL)
{
    int e = blockIdx.x * blockDim.x + threadIdx.x;
    if (e >= N_TOT * 64) return;
    int n = e >> 6, cc = e & 63;
    float v = 0.0f;
    if      (cc < 5)  v = la[n * 5 + cc];
    else if (cc == 5) v = lr[n];
    else if (cc < 36) {
        int w = (cc - 6) / 10, vv = (cc - 6) % 10;
        v = (msg[n * 3 + w] == vv) ? 1.0f : 0.0f;
    }
    else if (cc == 36) v = ct[n];
    else if (cc == 37) v = tte[n];
    XL[(size_t)n * LIN_IN_PAD + 512 + cc] = (f16)v;
}

// --------------------------- LSTM state kernels ----------------------------
__global__ void init_state(const float* __restrict__ h0, const float* __restrict__ c0,
                           float* __restrict__ hS, float* __restrict__ cS)
{
    int e = blockIdx.x * blockDim.x + threadIdx.x;
    if (e >= B_BATCH * HID) return;
    hS[e] = h0[e];
    cS[e] = c0[e];
}

__global__ void lstm_premask(const float* __restrict__ hS, const int* __restrict__ done,
                             int t, f16* __restrict__ hm)
{
    int e = blockIdx.x * blockDim.x + threadIdx.x;
    if (e >= B_BATCH * HID) return;
    int b = e / HID;
    float mask = 1.0f - (float)done[t * B_BATCH + b];
    hm[e] = (f16)(hS[e] * mask);
}

__global__ void lstm_gates(const float* __restrict__ GI, const float* __restrict__ GH,
                           const float* __restrict__ bih, const float* __restrict__ bhh,
                           const int* __restrict__ done, int t,
                           float* __restrict__ hS, float* __restrict__ cS,
                           f16* __restrict__ Hseq)
{
    int e = blockIdx.x * blockDim.x + threadIdx.x;
    if (e >= B_BATCH * HID) return;
    int b = e / HID, k = e - b * HID;
    int row = t * B_BATCH + b;
    size_t gi_base = (size_t)row * 2048;
    size_t gh_base = (size_t)b * 2048;
    float gi = GI[gi_base +         k] + GH[gh_base +         k] + bih[k]        + bhh[k];
    float gf = GI[gi_base +  512 + k] + GH[gh_base +  512 + k] + bih[512 + k]   + bhh[512 + k];
    float gg = GI[gi_base + 1024 + k] + GH[gh_base + 1024 + k] + bih[1024 + k]  + bhh[1024 + k];
    float go = GI[gi_base + 1536 + k] + GH[gh_base + 1536 + k] + bih[1536 + k]  + bhh[1536 + k];
    float mask  = 1.0f - (float)done[row];
    float cprev = cS[e] * mask;
    float cn = sigm(gf) * cprev + sigm(gi) * tanhf(gg);
    float hn = sigm(go) * tanhf(cn);
    cS[e] = cn;
    hS[e] = hn;
    Hseq[(size_t)row * HID + k] = (f16)hn;
}

// ------------------------------ final assembly -----------------------------
__global__ void assemble(const float* __restrict__ Ca, const float* __restrict__ Cv,
                         const float* __restrict__ ab3, const float* __restrict__ cb3,
                         float* __restrict__ out)
{
    int e = blockIdx.x * blockDim.x + threadIdx.x;
    if (e >= N_TOT * 33) return;
    int n = e / 33, j = e - n * 33;
    out[e] = (j < 32) ? (Ca[(size_t)n * 32 + j] + ab3[j])
                      : (Cv[(size_t)n * 32] + cb3[0]);
}

// ---------------------------------------------------------------------------
static inline size_t align256(size_t x) { return (x + 255) & ~(size_t)255; }

extern "C" void kernel_launch(void* const* d_in, const int* in_sizes, int n_in,
                              void* d_out, int out_size, void* d_ws, size_t ws_size,
                              hipStream_t stream)
{
    (void)in_sizes; (void)n_in; (void)out_size; (void)ws_size;

    const float* x   = (const float*)d_in[0];
    const int*   dn  = (const int*)  d_in[1];
    const int*   msg = (const int*)  d_in[2];
    const float* la  = (const float*)d_in[3];
    const float* lr  = (const float*)d_in[4];
    const float* ct  = (const float*)d_in[5];
    const float* tte = (const float*)d_in[6];
    const float* h0  = (const float*)d_in[7];
    const float* c0  = (const float*)d_in[8];
    const float* w1  = (const float*)d_in[9];  const float* b1  = (const float*)d_in[10];
    const float* w2  = (const float*)d_in[11]; const float* b2  = (const float*)d_in[12];
    const float* w3  = (const float*)d_in[13]; const float* b3  = (const float*)d_in[14];
    const float* lw  = (const float*)d_in[15]; const float* lb  = (const float*)d_in[16];
    const float* wih = (const float*)d_in[17]; const float* whh = (const float*)d_in[18];
    const float* bih = (const float*)d_in[19]; const float* bhh = (const float*)d_in[20];
    const float* cw1 = (const float*)d_in[21]; const float* cb1 = (const float*)d_in[22];
    const float* cw2 = (const float*)d_in[23]; const float* cb2 = (const float*)d_in[24];
    const float* cw3 = (const float*)d_in[25]; const float* cb3 = (const float*)d_in[26];
    const float* aw1 = (const float*)d_in[27]; const float* ab1 = (const float*)d_in[28];
    const float* aw2 = (const float*)d_in[29]; const float* ab2 = (const float*)d_in[30];
    const float* aw3 = (const float*)d_in[31]; const float* ab3 = (const float*)d_in[32];
    float* out = (float*)d_out;

    // ------------------------- workspace carve-out -------------------------
    char* ws = (char*)d_ws;
    size_t off = 0;
    auto alloc = [&](size_t bytes) -> char* {
        char* p = ws + off; off = align256(off + bytes); return p;
    };
    f16*   W1f  = (f16*)alloc((size_t)32   * 192  * 2);
    f16*   W2f  = (f16*)alloc((size_t)64   * 512  * 2);
    f16*   W3f  = (f16*)alloc((size_t)64   * 576  * 2);
    f16*   LWf  = (f16*)alloc((size_t)512  * 3136 * 2);
    f16*   WIHf = (f16*)alloc((size_t)2048 * 576  * 2);
    f16*   WHHf = (f16*)alloc((size_t)2048 * 512  * 2);
    f16*   CW1f = (f16*)alloc((size_t)256  * 512  * 2);
    f16*   CW2f = (f16*)alloc((size_t)256  * 256  * 2);
    f16*   CW3f = (f16*)alloc((size_t)32   * 256  * 2);   // cw3 padded 1 -> 32 rows
    f16*   AW1f = (f16*)alloc((size_t)256  * 512  * 2);
    f16*   AW2f = (f16*)alloc((size_t)256  * 256  * 2);
    f16*   AW3f = (f16*)alloc((size_t)32   * 256  * 2);
    f16*   Abuf = (f16*)alloc((size_t)256 * 81 * 512 * 2);     // max im2col chunk (~21 MB)
    f16*   Y1   = (f16*)alloc((size_t)N_TOT * 32 * 400 * 2);   // conv1 out
    f16*   Y2   = (f16*)alloc((size_t)N_TOT * 64 * 81  * 2);   // conv2 out
    f16*   A4   = (f16*)alloc((size_t)N_TOT * 3136 * 2);       // conv3 out == linear A
    f16*   XL   = (f16*)alloc((size_t)N_TOT * LIN_IN_PAD * 2); // lstm input matrix
    float* GI   = (float*)alloc((size_t)N_TOT * 2048 * 4);     // input gate projections
    float* GH   = (float*)alloc((size_t)B_BATCH * 2048 * 4);   // recurrent projections
    float* hS   = (float*)alloc((size_t)B_BATCH * HID * 4);
    float* cS   = (float*)alloc((size_t)B_BATCH * HID * 4);
    f16*   hm   = (f16*)alloc((size_t)B_BATCH * HID * 2);
    f16*   Hseq = (f16*)alloc((size_t)N_TOT * HID * 2);
    f16*   V1   = (f16*)alloc((size_t)N_TOT * 256 * 2);
    f16*   V2   = (f16*)alloc((size_t)N_TOT * 256 * 2);
    float* Cv   = (float*)alloc((size_t)N_TOT * 32 * 4);
    float* Ca   = (float*)alloc((size_t)N_TOT * 32 * 4);

    auto castw = [&](const float* src, f16* dst, int O, int K, int Opad, int Kp) {
        int n = Opad * Kp;
        hipLaunchKernelGGL(cast_pad, dim3((n + 255) / 256), dim3(256), 0, stream,
                           src, dst, O, K, Opad, Kp);
    };
    auto gemm = [&](const f16* A, int lda, const f16* B, int ldb,
                    int M, int Nc, int K, const float* bias, int act,
                    float* oF, f16* oH, int ldc, int coff, int SP, int NCtot) {
        dim3 grid((M + 32 * WPB - 1) / (32 * WPB), Nc / 32);
        hipLaunchKernelGGL(gemm_wmma_f16, grid, dim3(256), 0, stream,
                           A, lda, B, ldb, M, Nc, K, bias, act, oF, oH,
                           ldc, coff, SP, NCtot);
    };

    // ---- stage weights to f16 (padded where needed) ----
    castw(w1,  W1f,  32,   192,  32,   192);
    castw(w2,  W2f,  64,   512,  64,   512);
    castw(w3,  W3f,  64,   576,  64,   576);
    castw(lw,  LWf,  512,  3136, 512,  3136);
    castw(wih, WIHf, 2048, 550,  2048, 576);
    castw(whh, WHHf, 2048, 512,  2048, 512);
    castw(cw1, CW1f, 256,  512,  256,  512);
    castw(cw2, CW2f, 256,  256,  256,  256);
    castw(cw3, CW3f, 1,    256,  32,   256);
    castw(aw1, AW1f, 256,  512,  256,  512);
    castw(aw2, AW2f, 256,  256,  256,  256);
    castw(aw3, AW3f, 32,   256,  32,   256);

    // ---- aux columns of XL + LSTM state init ----
    hipLaunchKernelGGL(fill_aux, dim3((N_TOT * 64 + 255) / 256), dim3(256), 0, stream,
                       la, lr, msg, ct, tte, XL);
    hipLaunchKernelGGL(init_state, dim3((B_BATCH * HID + 255) / 256), dim3(256), 0, stream,
                       h0, c0, hS, cS);

    // ---- conv1: 16 chunks of 128 samples ----
    for (int chk = 0; chk < 16; ++chk) {
        int n_base = chk * 128, Mc = 128 * 400, nel = Mc * 192;
        hipLaunchKernelGGL(im2col1, dim3((nel + 255) / 256), dim3(256), 0, stream,
                           x, Abuf, n_base, nel);
        gemm(Abuf, 192, W1f, 192, Mc, 32, 192, b1, 1,
             nullptr, Y1 + (size_t)n_base * 32 * 400, 0, 0, 400, 32);
    }
    // ---- conv2: 8 chunks of 256 samples ----
    for (int chk = 0; chk < 8; ++chk) {
        int n_base = chk * 256, Mc = 256 * 81, nel = Mc * 512;
        hipLaunchKernelGGL(im2col2, dim3((nel + 255) / 256), dim3(256), 0, stream,
                           Y1, Abuf, n_base, nel);
        gemm(Abuf, 512, W2f, 512, Mc, 64, 512, b2, 1,
             nullptr, Y2 + (size_t)n_base * 64 * 81, 0, 0, 81, 64);
    }
    // ---- conv3: 8 chunks of 256 samples (writes flattened [N,3136]) ----
    for (int chk = 0; chk < 8; ++chk) {
        int n_base = chk * 256, Mc = 256 * 49, nel = Mc * 576;
        hipLaunchKernelGGL(im2col3, dim3((nel + 255) / 256), dim3(256), 0, stream,
                           Y2, Abuf, n_base, nel);
        gemm(Abuf, 576, W3f, 576, Mc, 64, 576, b3, 1,
             nullptr, A4 + (size_t)n_base * 3136, 0, 0, 49, 64);
    }

    // ---- linear 3136 -> 512 (relu) into XL cols 0..511 ----
    gemm(A4, 3136, LWf, 3136, N_TOT, 512, 3136, lb, 1, nullptr, XL, LIN_IN_PAD, 0, 1, 0);

    // ---- precompute input gate projections GI = XL @ W_ih^T ----
    gemm(XL, LIN_IN_PAD, WIHf, LIN_IN_PAD, N_TOT, 2048, LIN_IN_PAD,
         nullptr, 0, GI, nullptr, 2048, 0, 1, 0);

    // ---- LSTM recurrence: only h @ W_hh^T is sequential ----
    for (int t = 0; t < T_STEPS; ++t) {
        hipLaunchKernelGGL(lstm_premask, dim3(256), dim3(256), 0, stream, hS, dn, t, hm);
        gemm(hm, HID, WHHf, HID, B_BATCH, 2048, HID, nullptr, 0, GH, nullptr, 2048, 0, 1, 0);
        hipLaunchKernelGGL(lstm_gates, dim3(256), dim3(256), 0, stream,
                           GI, GH, bih, bhh, dn, t, hS, cS, Hseq);
    }

    // ---- critic head ----
    gemm(Hseq, HID, CW1f, HID, N_TOT, 256, HID, cb1, 2, nullptr, V1, 256, 0, 1, 0);
    gemm(V1, 256, CW2f, 256, N_TOT, 256, 256, cb2, 2, nullptr, V2, 256, 0, 1, 0);
    gemm(V2, 256, CW3f, 256, N_TOT, 32, 256, nullptr, 0, Cv, nullptr, 32, 0, 1, 0);
    // ---- actor head (reuses V1/V2) ----
    gemm(Hseq, HID, AW1f, HID, N_TOT, 256, HID, ab1, 2, nullptr, V1, 256, 0, 1, 0);
    gemm(V1, 256, AW2f, 256, N_TOT, 256, 256, ab2, 2, nullptr, V2, 256, 0, 1, 0);
    gemm(V2, 256, AW3f, 256, N_TOT, 32, 256, nullptr, 0, Ca, nullptr, 32, 0, 1, 0);

    // ---- assemble [actor(32) | value(1)] ----
    hipLaunchKernelGGL(assemble, dim3((N_TOT * 33 + 255) / 256), dim3(256), 0, stream,
                       Ca, Cv, ab3, cb3, out);
}
